// GNN_59605556134076
// MI455X (gfx1250) — matrix operand order repaired and checked by reading, not compiled, not run on hardware
//
#include <hip/hip_runtime.h>
#include <math.h>

// ---------------------------------------------------------------------------
// Problem constants (from reference): B=256 graphs, 256 nodes/graph, C=128,
// 2048 edges/graph, 3 layers, DENSE=256.
// ---------------------------------------------------------------------------
#define BT    256
#define NPG   256
#define CC    128
#define NN    (BT * NPG)     // 65536 nodes
#define EE    (BT * 2048)    // 524288 edges
#define DENSE 256

typedef __attribute__((ext_vector_type(16))) __bf16 bf16x16;
typedef __attribute__((ext_vector_type(8)))  float  f32x8;

#define WMMA_BF16(A_, B_, C_) \
  __builtin_amdgcn_wmma_f32_16x16x32_bf16(false, (A_), false, (B_), (short)0, (C_), false, false)

// ---- fragment loaders (unguarded, 16B-aligned vector loads) ----------------
// A fragment: lane covers one contiguous run of 16 floats: p[0..15]
__device__ __forceinline__ void frag_a_fast(bf16x16& f, const float* __restrict__ p) {
  const float4* q = (const float4*)p;
  float4 r0 = q[0], r1 = q[1], r2 = q[2], r3 = q[3];
  f[0]=(__bf16)r0.x; f[1]=(__bf16)r0.y; f[2]=(__bf16)r0.z; f[3]=(__bf16)r0.w;
  f[4]=(__bf16)r1.x; f[5]=(__bf16)r1.y; f[6]=(__bf16)r1.z; f[7]=(__bf16)r1.w;
  f[8]=(__bf16)r2.x; f[9]=(__bf16)r2.y; f[10]=(__bf16)r2.z; f[11]=(__bf16)r2.w;
  f[12]=(__bf16)r3.x; f[13]=(__bf16)r3.y; f[14]=(__bf16)r3.z; f[15]=(__bf16)r3.w;
}
// Contiguous 16-float fragment (transposed-B / K-operand): p[0..15]
__device__ __forceinline__ void frag_c16_fast(bf16x16& f, const float* __restrict__ p) {
  const float4* q = (const float4*)p;
  float4 r0 = q[0], r1 = q[1], r2 = q[2], r3 = q[3];
  f[0]=(__bf16)r0.x; f[1]=(__bf16)r0.y; f[2]=(__bf16)r0.z; f[3]=(__bf16)r0.w;
  f[4]=(__bf16)r1.x; f[5]=(__bf16)r1.y; f[6]=(__bf16)r1.z; f[7]=(__bf16)r1.w;
  f[8]=(__bf16)r2.x; f[9]=(__bf16)r2.y; f[10]=(__bf16)r2.z; f[11]=(__bf16)r2.w;
  f[12]=(__bf16)r3.x; f[13]=(__bf16)r3.y; f[14]=(__bf16)r3.z; f[15]=(__bf16)r3.w;
}
// Strided (k-major) B column fragment: 16 independent b32 loads, one wait
__device__ __forceinline__ void frag_col_fast(bf16x16& f, const float* __restrict__ p, int ld) {
  float t[16];
#pragma unroll
  for (int e = 0; e < 16; ++e) t[e] = p[(size_t)e * ld];
#pragma unroll
  for (int e = 0; e < 16; ++e) f[e] = (__bf16)t[e];
}

// ---------------------------------------------------------------------------
// FAST GEMM: no guards. Requires M%16==0, N%16==0 (and %(16*NT)), K%32==0,
// ldA%4==0 and (TB ? ldB%4==0 : true).  D = outScale*(act(A@B + bias) + Res).
// One wave computes NT adjacent 16x16 tiles in N (one A fragment -> NT WMMAs).
// ---------------------------------------------------------------------------
template <int NT, int TB>
__global__ void __launch_bounds__(256) gemm_fast_k(
    const float* __restrict__ A, const float* __restrict__ Bw,
    const float* __restrict__ bias, const float* __restrict__ Res,
    float* __restrict__ D, int M, int N, int K,
    int ldA, int ldB, int ldD, int ldRes,
    float outScale, int act)
{
  const int wave = threadIdx.x >> 5;
  const int lane = threadIdx.x & 31;
  const int tilesN  = N >> 4;
  const int tilesNg = tilesN / NT;
  const int tile = blockIdx.x * 8 + wave;
  if (tile >= (M >> 4) * tilesNg) return;
  const int tm  = tile / tilesNg;
  const int tn0 = (tile % tilesNg) * NT;
  const int half = lane >> 4;
  const int l16  = lane & 15;
  const int row  = tm * 16 + l16;

  f32x8 acc[NT];
#pragma unroll
  for (int t = 0; t < NT; ++t) acc[t] = (f32x8){};

  const float* pA = A + (size_t)row * ldA + 16 * half;
  for (int k0 = 0; k0 < K; k0 += 32) {
    bf16x16 a;
    frag_a_fast(a, pA + k0);
#pragma unroll
    for (int t = 0; t < NT; ++t) {
      const int col = (tn0 + t) * 16 + l16;
      bf16x16 b;
      if (TB) frag_c16_fast(b, Bw + (size_t)col * ldB + k0 + 16 * half);
      else    frag_col_fast(b, Bw + (size_t)(k0 + 16 * half) * ldB + col, ldB);
      acc[t] = WMMA_BF16(a, b, acc[t]);
    }
  }
  // epilogue: C/D layout -> VGPR i: row i + 8*half, col l16
#pragma unroll
  for (int t = 0; t < NT; ++t) {
#pragma unroll
    for (int i = 0; i < 8; ++i) {
      int r = tm * 16 + i + 8 * half;
      int c = (tn0 + t) * 16 + l16;
      float v = acc[t][i];
      if (bias) v += bias[c];
      if (act == 1) v = v > 0.0f ? v : 0.0f;
      if (Res) v += Res[(size_t)r * ldRes + c];
      D[(size_t)r * ldD + c] = v * outScale;
    }
  }
}

// ---------------------------------------------------------------------------
// Generic guarded GEMM (only for tiny shapes: K=17/30, N=8/1, M=1/4).
// ---------------------------------------------------------------------------
__global__ void __launch_bounds__(256) gemm_gen_k(
    const float* __restrict__ A, const float* __restrict__ Bw,
    const float* __restrict__ bias, const float* __restrict__ Res,
    float* __restrict__ D, int M, int N, int K,
    int ldA, int ldB, int ldD, int ldRes,
    float outScale, int transB, int act)
{
  const int wave = threadIdx.x >> 5;
  const int lane = threadIdx.x & 31;
  const int tilesN = (N + 15) >> 4;
  const int tilesM = (M + 15) >> 4;
  const int tile = blockIdx.x * 8 + wave;
  if (tile >= tilesM * tilesN) return;
  const int tm = tile / tilesN, tn = tile % tilesN;
  const int half = lane >> 4;
  const int l16  = lane & 15;
  const int row  = tm * 16 + l16;
  const int col  = tn * 16 + l16;

  f32x8 acc = {};
  for (int k0 = 0; k0 < K; k0 += 32) {
    bf16x16 a, b;
#pragma unroll
    for (int e = 0; e < 16; ++e) {
      int k = k0 + 16 * half + e;
      float v = 0.0f;
      if (row < M && k < K) v = A[(size_t)row * ldA + k];
      a[e] = (__bf16)v;
    }
#pragma unroll
    for (int e = 0; e < 16; ++e) {
      int k = k0 + 16 * half + e;
      float v = 0.0f;
      if (col < N && k < K)
        v = transB ? Bw[(size_t)col * ldB + k] : Bw[(size_t)k * ldB + col];
      b[e] = (__bf16)v;
    }
    acc = WMMA_BF16(a, b, acc);
  }
#pragma unroll
  for (int i = 0; i < 8; ++i) {
    int r = tm * 16 + i + 8 * half;
    int c = tn * 16 + l16;
    if (r < M && c < N) {
      float v = acc[i];
      if (bias) v += bias[c];
      if (act == 1) v = v > 0.0f ? v : 0.0f;
      if (Res) v += Res[(size_t)r * ldRes + c];
      D[(size_t)r * ldD + c] = v * outScale;
    }
  }
}

// ---------------------------------------------------------------------------
// Attention softmax(Q K^T * scale) V, NKV=256 keys, head dim DH compile-time.
// One wave per (batch, head, 16-row query tile); score strip in LDS.
// FULLQ=false: nq==1 (PMA) -> guarded Q loads / output stores.
// ---------------------------------------------------------------------------
#define SROW (NPG + 4)   // 16B-aligned LDS row stride
template <int DH, bool FULLQ>
__global__ void __launch_bounds__(32) attn_fast_k(
    const float* __restrict__ Q, const float* __restrict__ Km,
    const float* __restrict__ V, float* __restrict__ O,
    int nq, int heads, int ldq, int ldkv, int ldo,
    long long qBatch, long long kvBatch, long long oBatch,
    float scale)
{
  __shared__ float S[16][SROW];
  const int lane = threadIdx.x & 31;
  const int mtiles = (nq + 15) >> 4;
  const int job = blockIdx.x;
  const int mt = job % mtiles;
  const int bh = job / mtiles;
  const int h  = bh % heads;
  const int b  = bh / heads;
  const float* Qb = Q + (size_t)b * qBatch + h * DH;
  const float* Kb = Km + (size_t)b * kvBatch + h * DH;
  const float* Vb = V + (size_t)b * kvBatch + h * DH;
  float* Ob       = O + (size_t)b * oBatch + h * DH;
  const int half = lane >> 4, l16 = lane & 15;
  const int qrow = mt * 16 + l16;

  // Phase 1: S = scale * Q K^T   (16 key tiles, DH/32 k-steps)
  for (int nt = 0; nt < NPG / 16; ++nt) {
    f32x8 acc = {};
    const int krow = nt * 16 + l16;
    for (int k0 = 0; k0 < DH; k0 += 32) {
      bf16x16 a, bf;
      if constexpr (FULLQ) {
        frag_a_fast(a, Qb + (size_t)qrow * ldq + k0 + 16 * half);
      } else {
#pragma unroll
        for (int e = 0; e < 16; ++e) {
          int k = k0 + 16 * half + e;
          float v = 0.0f;
          if (qrow < nq) v = Qb[(size_t)qrow * ldq + k];
          a[e] = (__bf16)v;
        }
      }
      frag_c16_fast(bf, Kb + (size_t)krow * ldkv + k0 + 16 * half);
      acc = WMMA_BF16(a, bf, acc);
    }
#pragma unroll
    for (int i = 0; i < 8; ++i)
      S[i + 8 * half][nt * 16 + l16] = acc[i] * scale;
  }
  __syncthreads();

  // Phase 2: row softmax (lanes 0..15, one row each)
  if (lane < 16 && (FULLQ || lane < nq)) {
    float mx = -3.0e38f;
    for (int j = 0; j < NPG; ++j) mx = fmaxf(mx, S[lane][j]);
    float sum = 0.0f;
    for (int j = 0; j < NPG; ++j) { float e = expf(S[lane][j] - mx); S[lane][j] = e; sum += e; }
    float inv = 1.0f / sum;
    for (int j = 0; j < NPG; ++j) S[lane][j] *= inv;
  }
  __syncthreads();

  // Phase 3: O = P @ V   (DH/16 col tiles, 8 k-steps over 256 keys)
  for (int nt = 0; nt < DH / 16; ++nt) {
    f32x8 acc = {};
    const int vcol = nt * 16 + l16;
    for (int k0 = 0; k0 < NPG; k0 += 32) {
      bf16x16 a, bf;
      const float* sp = &S[l16][k0 + 16 * half];
#pragma unroll
      for (int e = 0; e < 8; ++e)  a[e] = (__bf16)sp[e];
#pragma unroll
      for (int e = 8; e < 16; ++e) a[e] = (__bf16)sp[e];
      frag_col_fast(bf, Vb + (size_t)(k0 + 16 * half) * ldkv + vcol, ldkv);
      acc = WMMA_BF16(a, bf, acc);
    }
#pragma unroll
    for (int i = 0; i < 8; ++i) {
      int r = mt * 16 + i + 8 * half;
      if (FULLQ || r < nq) Ob[(size_t)r * ldo + nt * 16 + l16] = acc[i];
    }
  }
}

// ---------------------------------------------------------------------------
// GINE edge message + scatter:  agg[dst] += relu(x[src] + ea @ Wc + bc)
// (edge linear composed to a [4,C] weight on device -> 4 MACs/element)
// ---------------------------------------------------------------------------
__global__ void __launch_bounds__(256) gine_edge_k(
    const float* __restrict__ x, const float* __restrict__ ea,
    const float* __restrict__ Wc, const float* __restrict__ bc,
    const int* __restrict__ src, const int* __restrict__ dst,
    float* __restrict__ agg, int E, int C)
{
  int idx = blockIdx.x * blockDim.x + threadIdx.x;
  if (idx >= E * C) return;
  int e = idx / C, c = idx - e * C;
  float lin = bc[c];
#pragma unroll
  for (int j = 0; j < 4; ++j) lin += ea[(size_t)e * 4 + j] * Wc[j * C + c];
  float m = x[(size_t)src[e] * C + c] + lin;
  if (m > 0.0f) atomicAdd(&agg[(size_t)dst[e] * C + c], m);
}

// ---------------------------------------------------------------------------
// GraphNorm: per (graph, channel) mean/var over NPG nodes.
// ---------------------------------------------------------------------------
__global__ void __launch_bounds__(CC) graph_norm_k(
    const float* __restrict__ x, float* __restrict__ out,
    const float* __restrict__ w, const float* __restrict__ b,
    const float* __restrict__ ms)
{
  const int g = blockIdx.x, c = threadIdx.x;
  const float* xg = x + (size_t)g * NPG * CC;
  float mean = 0.0f;
  for (int n = 0; n < NPG; ++n) mean += xg[n * CC + c];
  mean *= (1.0f / NPG);
  float m = ms[c] * mean;
  float var = 0.0f;
  for (int n = 0; n < NPG; ++n) { float t = xg[n * CC + c] - m; var += t * t; }
  var *= (1.0f / NPG);
  float inv = rsqrtf(var + 1e-5f);
  float W = w[c], Bb = b[c];
  float* og = out + (size_t)g * NPG * CC;
  for (int n = 0; n < NPG; ++n)
    og[n * CC + c] = W * ((xg[n * CC + c] - m) * inv) + Bb;
}

// ---------------------------------------------------------------------------
// Small elementwise helpers
// ---------------------------------------------------------------------------
__global__ void fill_k(float* p, long long n, float v) {
  long long i = (long long)blockIdx.x * blockDim.x + threadIdx.x;
  if (i < n) p[i] = v;
}
__global__ void ew_add_scale_k(const float* __restrict__ a, const float* __restrict__ b,
                               float* __restrict__ o, long long n, float sc) {
  long long i = (long long)blockIdx.x * blockDim.x + threadIdx.x;
  if (i >= n) return;
  float v = a[i];
  if (b) v += b[i];
  o[i] = sc * v;
}
__global__ void copy2d_k(const float* __restrict__ in, float* __restrict__ out,
                         int M, int Ccols, int ldI, int ldO) {
  long long i = (long long)blockIdx.x * blockDim.x + threadIdx.x;
  if (i >= (long long)M * Ccols) return;
  int r = (int)(i / Ccols), c = (int)(i - (long long)r * Ccols);
  out[(size_t)r * ldO + c] = in[(size_t)r * ldI + c];
}

// ---------------------------------------------------------------------------
// Host-side orchestration
// ---------------------------------------------------------------------------
struct Lin { const float *b, *w; };
struct Mha { const float *in_b, *in_w, *out_b, *out_w; };
struct Mab { Mha attn; Lin lin; };
struct SetAggP { Mab dec; Mab enc; Lin pma_lin; Mab pma_mab; const float* seed; };
struct GnP { const float *bias, *ms, *weight; };
struct LayerP { SetAggP aggr; Mha attn; Lin lin_edge, nn1, nn2; GnP gn; Lin mlp1, mlp2; };

static void gemm(hipStream_t st, const float* A, const float* Bw, const float* bias,
                 const float* Res, float* D, int M, int N, int K,
                 int ldA, int ldB, int ldD, int ldRes,
                 float outScale, int transB, int act) {
  const bool fast = (M % 16 == 0) && (N % 16 == 0) && (K % 32 == 0) &&
                    ((ldA & 3) == 0) && (!transB || (ldB & 3) == 0);
  int tilesM = (M + 15) / 16, tilesN = (N + 15) / 16;
  if (fast) {
    if ((tilesN & 3) == 0) {
      int blocks = (tilesM * (tilesN / 4) + 7) / 8;
      if (transB) gemm_fast_k<4, 1><<<blocks, 256, 0, st>>>(A, Bw, bias, Res, D, M, N, K, ldA, ldB, ldD, ldRes, outScale, act);
      else        gemm_fast_k<4, 0><<<blocks, 256, 0, st>>>(A, Bw, bias, Res, D, M, N, K, ldA, ldB, ldD, ldRes, outScale, act);
    } else {
      int blocks = (tilesM * tilesN + 7) / 8;
      if (transB) gemm_fast_k<1, 1><<<blocks, 256, 0, st>>>(A, Bw, bias, Res, D, M, N, K, ldA, ldB, ldD, ldRes, outScale, act);
      else        gemm_fast_k<1, 0><<<blocks, 256, 0, st>>>(A, Bw, bias, Res, D, M, N, K, ldA, ldB, ldD, ldRes, outScale, act);
    }
  } else {
    int blocks = (tilesM * tilesN + 7) / 8;
    gemm_gen_k<<<blocks, 256, 0, st>>>(A, Bw, bias, Res, D, M, N, K, ldA, ldB, ldD, ldRes, outScale, transB, act);
  }
}
static void ew(hipStream_t st, const float* a, const float* b, float* o, long long n, float sc) {
  ew_add_scale_k<<<(int)((n + 255) / 256), 256, 0, st>>>(a, b, o, n, sc);
}

struct Scr {
  float *h, *agg, *nb0, *nb1, *nb2, *QKV, *KV, *t2c;
  float *g, *gtmp, *sb0, *sb1, *sb2, *sqkv, *qseed, *Wc, *bc, *hb1, *hb2, *hb3;
};

// SetTransformerAggregation: X [BT, NPG, CC] -> out [BT, CC]
static void run_set_agg(hipStream_t st, const SetAggP& P, const float* X, float* gout, Scr& S) {
  const float sc128 = 1.0f / sqrtf((float)CC);
  // --- encoder SAB ---
  gemm(st, X, P.enc.attn.in_w, P.enc.attn.in_b, nullptr, S.QKV,
       NN, 3 * CC, CC, CC, CC, 3 * CC, 0, 1.0f, 1, 0);
  attn_fast_k<CC, true><<<BT * (NPG / 16), 32, 0, st>>>(
      S.QKV, S.QKV + CC, S.QKV + 2 * CC, S.nb0, NPG, 1, 3 * CC, 3 * CC, CC,
      (long long)NPG * 3 * CC, (long long)NPG * 3 * CC, (long long)NPG * CC, sc128);
  gemm(st, S.nb0, P.enc.attn.out_w, P.enc.attn.out_b, X, S.nb1,
       NN, CC, CC, CC, CC, CC, CC, 1.0f, 1, 0);               // out = mha + x
  gemm(st, S.nb1, P.enc.lin.w, P.enc.lin.b, S.nb1, S.nb0,
       NN, CC, CC, CC, CC, CC, CC, 1.0f, 0, 1);               // out += relu(lin(out))
  // --- hp = relu(pma_lin(h)) ---
  gemm(st, S.nb0, P.pma_lin.w, P.pma_lin.b, nullptr, S.nb1,
       NN, CC, CC, CC, CC, CC, 0, 1.0f, 0, 1);
  // --- PMA: Q from seed (shared), KV from hp ---
  gemm(st, P.seed, P.pma_mab.attn.in_w, P.pma_mab.attn.in_b, nullptr, S.qseed,
       1, CC, CC, CC, CC, CC, 0, 1.0f, 1, 0);
  gemm(st, S.nb1, P.pma_mab.attn.in_w + CC * CC, P.pma_mab.attn.in_b + CC, nullptr, S.KV,
       NN, 2 * CC, CC, CC, CC, 2 * CC, 0, 1.0f, 1, 0);
  attn_fast_k<CC, false><<<BT, 32, 0, st>>>(
      S.qseed, S.KV, S.KV + CC, S.sb0, 1, 1, CC, 2 * CC, CC,
      0LL, (long long)NPG * 2 * CC, (long long)CC, sc128);
  gemm(st, S.sb0, P.pma_mab.attn.out_w, P.pma_mab.attn.out_b, P.seed, S.sb1,
       BT, CC, CC, CC, CC, CC, 0 /*broadcast seed*/, 1.0f, 1, 0);
  gemm(st, S.sb1, P.pma_mab.lin.w, P.pma_mab.lin.b, S.sb1, S.sb2,
       BT, CC, CC, CC, CC, CC, CC, 1.0f, 0, 1);
  // --- decoder SAB on [BT,1,CC]: 1x1 attention => softmax(scalar)=1 => O=V ---
  gemm(st, S.sb2, P.dec.attn.in_w, P.dec.attn.in_b, nullptr, S.sqkv,
       BT, 3 * CC, CC, CC, CC, 3 * CC, 0, 1.0f, 1, 0);
  copy2d_k<<<(BT * CC + 255) / 256, 256, 0, st>>>(S.sqkv + 2 * CC, S.sb0, BT, CC, 3 * CC, CC);
  gemm(st, S.sb0, P.dec.attn.out_w, P.dec.attn.out_b, S.sb2, S.sb1,
       BT, CC, CC, CC, CC, CC, CC, 1.0f, 1, 0);
  gemm(st, S.sb1, P.dec.lin.w, P.dec.lin.b, S.sb1, gout,
       BT, CC, CC, CC, CC, CC, CC, 1.0f, 0, 1);
}

extern "C" void kernel_launch(void* const* d_in, const int* in_sizes, int n_in,
                              void* d_out, int out_size, void* d_ws, size_t ws_size,
                              hipStream_t stream) {
  if (n_in < 155) return;  // 6 tensors + 149 param leaves expected
  const float BN = 1.0f / sqrtf(1.0f + 1e-5f);

  const float* x_in  = (const float*)d_in[0];   // [NN, 9]
  const float* pe    = (const float*)d_in[1];   // [NN, 30]
  const float* ea    = (const float*)d_in[2];   // [EE, 4]
  const int*   eidx  = (const int*)d_in[3];     // [2, EE]
  const int*   src   = eidx;
  const int*   dst   = eidx + EE;
  // d_in[4]=batch_index, d_in[5]=gf : unused by the reference forward.

  // ---- parse params (jax tree_leaves: dict keys sorted alphabetically) ----
  int pi = 6;
  auto nf    = [&]() { return (const float*)d_in[pi++]; };
  auto rdLin = [&]() { Lin l; l.b = nf(); l.w = nf(); return l; };
  auto rdMha = [&]() { Mha m; m.in_b = nf(); m.in_w = nf(); m.out_b = nf(); m.out_w = nf(); return m; };
  auto rdMab = [&]() { Mab m; m.attn = rdMha(); m.lin = rdLin(); return m; };
  auto rdSet = [&]() { SetAggP s; s.dec = rdMab(); s.enc = rdMab();
                       s.pma_lin = rdLin(); s.pma_mab = rdMab(); s.seed = nf(); return s; };
  SetAggP aggr0 = rdSet();
  Lin edge_emb  = rdLin();
  Lin hl1 = rdLin(), hl2 = rdLin(), hl3 = rdLin(), hl4 = rdLin();
  LayerP L[3];
  for (int i = 0; i < 3; ++i) {
    L[i].aggr = rdSet(); L[i].attn = rdMha();
    L[i].lin_edge = rdLin(); L[i].nn1 = rdLin(); L[i].nn2 = rdLin();
    L[i].gn.bias = nf(); L[i].gn.ms = nf(); L[i].gn.weight = nf();
    L[i].mlp1 = rdLin(); L[i].mlp2 = rdLin();
  }
  Lin node_emb = rdLin();
  Lin pe_lin   = rdLin();

  // ---- scratch carve-out (~400 MB of f32) ----
  float* W = (float*)d_ws;
  size_t off = 0;
  auto al = [&](size_t n) { float* p = W + off; off += n; return p; };
  Scr S;
  S.h    = al((size_t)NN * CC);
  S.agg  = al((size_t)NN * CC);
  S.nb0  = al((size_t)NN * CC);
  S.nb1  = al((size_t)NN * CC);
  S.nb2  = al((size_t)NN * CC);
  S.QKV  = al((size_t)NN * 3 * CC);
  S.KV   = al((size_t)NN * 2 * CC);
  S.t2c  = al((size_t)NN * 2 * CC);
  S.g    = al(BT * CC);
  S.gtmp = al(BT * CC);
  S.sb0  = al(BT * CC);
  S.sb1  = al(BT * CC);
  S.sb2  = al(BT * CC);
  S.sqkv = al(BT * 3 * CC);
  S.qseed= al(CC);
  S.Wc   = al(4 * CC);
  S.bc   = al(CC);
  S.hb1  = al(BT * DENSE);
  S.hb2  = al(BT * (DENSE / 2));
  S.hb3  = al(BT * (DENSE / 4));
  float* peS = S.t2c;   // alias: [NN,30] fits in [NN,256], used only in prologue
  float* xin = S.KV;    // alias: [NN,17] fits in [NN,256], used only in prologue

  // ---- prologue: xin = concat(x, pe_lin(pe*BN)); h = node_emb(xin) ----
  ew(stream, pe, nullptr, peS, (long long)NN * 30, BN);
  copy2d_k<<<(int)(((long long)NN * 9 + 255) / 256), 256, 0, stream>>>(x_in, xin, NN, 9, 9, 17);
  gemm(stream, peS, pe_lin.w, pe_lin.b, nullptr, xin + 9,
       NN, 8, 30, 30, 8, 17, 0, 1.0f, 0, 0);
  gemm(stream, xin, node_emb.w, node_emb.b, nullptr, S.h,
       NN, CC, 17, 17, CC, CC, 0, 1.0f, 0, 0);

  // ---- g = set_agg(aggr0, h) ----
  run_set_agg(stream, aggr0, S.h, S.g, S);

  // ---- GPS layers ----
  const float sc32 = 1.0f / sqrtf(32.0f);
  for (int li = 0; li < 3; ++li) {
    const LayerP& lp = L[li];
    // GINE: compose edge_emb with gine_lin_edge -> Wc[4,CC], bc[CC]
    fill_k<<<(int)(((long long)NN * CC + 255) / 256), 256, 0, stream>>>(S.agg, (long long)NN * CC, 0.0f);
    gemm(stream, edge_emb.w, lp.lin_edge.w, nullptr, nullptr, S.Wc,
         4, CC, CC, CC, CC, CC, 0, 1.0f, 0, 0);
    gemm(stream, edge_emb.b, lp.lin_edge.w, lp.lin_edge.b, nullptr, S.bc,
         1, CC, CC, CC, CC, CC, 0, 1.0f, 0, 0);
    gine_edge_k<<<(EE * CC) / 256, 256, 0, stream>>>(S.h, ea, S.Wc, S.bc, src, dst, S.agg, EE, CC);
    ew(stream, S.h, S.agg, S.nb0, (long long)NN * CC, 1.0f);                 // x + agg
    gemm(stream, S.nb0, lp.nn1.w, lp.nn1.b, nullptr, S.nb1,
         NN, CC, CC, CC, CC, CC, 0, 1.0f, 0, 1);                             // relu(nn1)
    gemm(stream, S.nb1, lp.nn2.w, lp.nn2.b, S.h, S.nb0,
         NN, CC, CC, CC, CC, CC, CC, BN, 0, 0);                              // h1=(gine+x)*BN
    // local MHA (heads=4, d=32)
    gemm(stream, S.h, lp.attn.in_w, lp.attn.in_b, nullptr, S.QKV,
         NN, 3 * CC, CC, CC, CC, 3 * CC, 0, 1.0f, 1, 0);
    attn_fast_k<32, true><<<BT * 4 * (NPG / 16), 32, 0, stream>>>(
        S.QKV, S.QKV + CC, S.QKV + 2 * CC, S.nb1, NPG, 4, 3 * CC, 3 * CC, CC,
        (long long)NPG * 3 * CC, (long long)NPG * 3 * CC, (long long)NPG * CC, sc32);
    gemm(stream, S.nb1, lp.attn.out_w, lp.attn.out_b, S.h, S.nb2,
         NN, CC, CC, CC, CC, CC, CC, BN, 1, 0);                              // h2=(a+x)*BN
    ew(stream, S.nb0, S.nb2, S.nb1, (long long)NN * CC, 1.0f);               // out=h1+h2
    gemm(stream, S.nb1, lp.mlp1.w, lp.mlp1.b, nullptr, S.t2c,
         NN, 2 * CC, CC, CC, 2 * CC, 2 * CC, 0, 1.0f, 0, 1);
    gemm(stream, S.t2c, lp.mlp2.w, lp.mlp2.b, S.nb1, S.nb2,
         NN, CC, 2 * CC, 2 * CC, CC, CC, CC, BN, 0, 0);                      // (out+mlp)*BN
    graph_norm_k<<<BT, CC, 0, stream>>>(S.nb2, S.h, lp.gn.weight, lp.gn.bias, lp.gn.ms);
    // g += set_agg(layer aggr, h)
    run_set_agg(stream, lp.aggr, S.h, S.gtmp, S);
    ew(stream, S.g, S.gtmp, S.g, (long long)BT * CC, 1.0f);
  }

  // ---- head ----
  gemm(stream, S.g,   hl1.w, hl1.b, nullptr, S.hb1, BT, DENSE,     CC,        CC,        DENSE,     DENSE,     0, 1.0f, 0, 1);
  gemm(stream, S.hb1, hl2.w, hl2.b, nullptr, S.hb2, BT, DENSE / 2, DENSE,     DENSE,     DENSE / 2, DENSE / 2, 0, 1.0f, 0, 1);
  gemm(stream, S.hb2, hl3.w, hl3.b, nullptr, S.hb3, BT, DENSE / 4, DENSE / 2, DENSE / 2, DENSE / 4, DENSE / 4, 0, 1.0f, 0, 1);
  gemm(stream, S.hb3, hl4.w, hl4.b, nullptr, (float*)d_out, BT, 1, DENSE / 4, DENSE / 4, 1, 1, 0, 1.0f, 0, 0);
}
